// GNN_79310866088342
// MI455X (gfx1250) — matrix operand order repaired and checked by reading, not compiled, not run on hardware
//
#include <hip/hip_runtime.h>

// ---------------------------------------------------------------------------
// Types for CDNA5 WMMA (wave32): v16bf A/B fragments, v8f C/D accumulator.
// ---------------------------------------------------------------------------
typedef __attribute__((ext_vector_type(16))) __bf16 v16bf;
typedef __attribute__((ext_vector_type(8)))  __bf16 v8bf;
typedef __attribute__((ext_vector_type(8)))  float  v8f;
typedef __attribute__((ext_vector_type(8)))  float  f32x8;

// ---------------------------------------------------------------------------
// 0) fp32 -> bf16 weight conversion WITH transpose: w[l][k][n] -> o[l][n][k]
//    (N-major storage makes every WMMA B fragment two contiguous 16-byte runs)
// ---------------------------------------------------------------------------
__global__ void k_cvtT(const float* __restrict__ w, __bf16* __restrict__ o,
                       int K, int Nc, long total) {
    long i = (long)blockIdx.x * blockDim.x + threadIdx.x;
    if (i >= total) return;
    long kn = (long)K * Nc;
    int  l  = (int)(i / kn);
    int  r  = (int)(i % kn);
    int  k  = r / Nc;
    int  n  = r % Nc;
    o[((size_t)l * Nc + n) * K + k] = (__bf16)w[i];
}

// ---------------------------------------------------------------------------
// 1) node embedding: h[i] = x_emb1[x[i,0]] + x_emb2[x[i,1]]   (float4 lanes)
// ---------------------------------------------------------------------------
__global__ void k_embed(const int* __restrict__ x,
                        const float* __restrict__ e1, const float* __restrict__ e2,
                        float* __restrict__ h, int Nn) {
    int idx = blockIdx.x * blockDim.x + threadIdx.x;   // Nn*32 float4 slots
    int node = idx >> 5, c = idx & 31;
    if (node >= Nn) return;
    const float4* a = (const float4*)(e1 + (size_t)x[node * 2 + 0] * 128);
    const float4* b = (const float4*)(e2 + (size_t)x[node * 2 + 1] * 128);
    float4 va = a[c], vb = b[c];
    float4 o; o.x = va.x + vb.x; o.y = va.y + vb.y; o.z = va.z + vb.z; o.w = va.w + vb.w;
    ((float4*)h)[(size_t)node * 32 + c] = o;
}

// ---------------------------------------------------------------------------
// 2) zero agg (N*128 f32) and stats (256 f32)
// ---------------------------------------------------------------------------
__global__ void k_zero(float4* __restrict__ agg4, float4* __restrict__ stats4, long n4) {
    long i = (long)blockIdx.x * blockDim.x + threadIdx.x;
    float4 z; z.x = 0.f; z.y = 0.f; z.z = 0.f; z.w = 0.f;
    if (i < n4) agg4[i] = z;
    if (i < 64) stats4[i] = z;
}

// ---------------------------------------------------------------------------
// 3) message + scatter-add. One wave per edge (incl. self loops at e >= E):
//    agg[dst] += h[src] + edge_emb1[ea0] + edge_emb2[ea1]
// ---------------------------------------------------------------------------
__global__ __launch_bounds__(256) void k_msg(const int* __restrict__ edge_index,
                                             const int* __restrict__ edge_attr,
                                             const float* __restrict__ h,
                                             const float* __restrict__ e1,  // [6,128]
                                             const float* __restrict__ e2,  // [3,128]
                                             float* __restrict__ agg,
                                             int E, int Nn) {
    int lane = threadIdx.x & 31;
    long e = (long)blockIdx.x * (blockDim.x >> 5) + (threadIdx.x >> 5);
    long total = (long)E + Nn;
    if (e >= total) return;
    int s, d, a0, a1;
    if (e < E) {
        s  = edge_index[e];
        d  = edge_index[(long)E + e];
        a0 = edge_attr[e * 2 + 0];
        a1 = edge_attr[e * 2 + 1];
    } else {
        s = d = (int)(e - E);
        a0 = 4; a1 = 0;                       // self-loop attr
    }
    const float4* hs  = (const float4*)(h  + (size_t)s  * 128);
    const float4* w1r = (const float4*)(e1 + (size_t)a0 * 128);
    const float4* w2r = (const float4*)(e2 + (size_t)a1 * 128);
    float4 vh = hs[lane], v1 = w1r[lane], v2 = w2r[lane];
    float v[4];
    v[0] = vh.x + v1.x + v2.x; v[1] = vh.y + v1.y + v2.y;
    v[2] = vh.z + v1.z + v2.z; v[3] = vh.w + v1.w + v2.w;
    float* base = agg + (size_t)d * 128 + lane * 4;
#pragma unroll
    for (int j = 0; j < 4; ++j) atomicAdd(base + j, v[j]);
}

// ---------------------------------------------------------------------------
// WMMA fragment index map (ISA 7.12.2, wave32, 16-bit data):
//   A 16x32: lane=16*half+m, elem e -> K = kk + 8*half + (e<8 ? e : 8+e)
//            == two contiguous 8-elem runs at kk+8*half and kk+16+8*half
//   B 32x16: lane=16*half+n, elem e -> K = kk + 8*half + (e<8 ? e : 8+e)
//            == two 8-elem runs of column n at kk+8*half and kk+16+8*half (transposed storage!)
//   C 16x16 f32: VGPR r -> M = r + 8*half, N = lane&15
// ---------------------------------------------------------------------------
#define SHUF16(lo, hi) __builtin_shufflevector(lo, hi, 0,1,2,3,4,5,6,7,8,9,10,11,12,13,14,15)

// 4) GEMM1: hid = relu(agg[f32, Nx128] @ w1 + b1) -> bf16 [Nx256]
//    w1bT: bf16 transposed [256][128].  8 waves: 2 row-tiles x 4 col-tiles.
__global__ __launch_bounds__(256) void k_gemm1(const float* __restrict__ A,
                                               const __bf16* __restrict__ BT,
                                               const float* __restrict__ bias,
                                               __bf16* __restrict__ Out, int M) {
    int tid = threadIdx.x, wave = tid >> 5, lane = tid & 31;
    int half = lane >> 4, lm = lane & 15;
    int r0 = blockIdx.x * 32 + (wave >> 2) * 16;
    int c0 = (wave & 3) * 64;
    v8f acc[4] = {};
    int ar = r0 + lm; if (ar > M - 1) ar = M - 1;        // clamp A loads
    const float* arowBase = A + (size_t)ar * 128 + 8 * half;
#pragma unroll
    for (int ks = 0; ks < 4; ++ks) {
        int kk = ks * 32;
        // A: two aligned float8 runs, hardware-converted to bf16
        f32x8 flo = *(const f32x8*)(arowBase + kk);
        f32x8 fhi = *(const f32x8*)(arowBase + kk + 16);
        v8bf alo = __builtin_convertvector(flo, v8bf);
        v8bf ahi = __builtin_convertvector(fhi, v8bf);
        v16bf a = SHUF16(alo, ahi);
#pragma unroll
        for (int j = 0; j < 4; ++j) {
            int n = c0 + 16 * j + lm;
            // B: two aligned 16-byte runs of column n at kk+8*half and kk+16+8*half
            const __bf16* bcol = BT + (size_t)n * 128 + kk + 8 * half;
            v16bf b = SHUF16(*(const v8bf*)bcol, *(const v8bf*)(bcol + 16));
            acc[j] = __builtin_amdgcn_wmma_f32_16x16x32_bf16(
                false, a, false, b, (short)0, acc[j], false, false);
        }
    }
#pragma unroll
    for (int j = 0; j < 4; ++j) {
        int n = c0 + 16 * j + lm;
        float bn = bias[n];
#pragma unroll
        for (int r = 0; r < 8; ++r) {
            int row = r0 + r + 8 * half;
            if (row < M) {
                float v = acc[j][r] + bn;
                Out[(size_t)row * 256 + n] = (__bf16)fmaxf(v, 0.0f);
            }
        }
    }
}

// 5) GEMM2: hn = hid[bf16, Nx256] @ w2 + b2 -> f32 [Nx128]
//    w2bT: bf16 transposed [128][256].  8 waves: 4 row-tiles x 2 col-tiles.
__global__ __launch_bounds__(256) void k_gemm2(const __bf16* __restrict__ A,
                                               const __bf16* __restrict__ BT,
                                               const float* __restrict__ bias,
                                               float* __restrict__ Out, int M) {
    int tid = threadIdx.x, wave = tid >> 5, lane = tid & 31;
    int half = lane >> 4, lm = lane & 15;
    int r0 = blockIdx.x * 64 + (wave >> 1) * 16;
    int c0 = (wave & 1) * 64;
    v8f acc[4] = {};
    int ar = r0 + lm; if (ar > M - 1) ar = M - 1;
    const __bf16* arowBase = A + (size_t)ar * 256 + 8 * half;
#pragma unroll
    for (int ks = 0; ks < 8; ++ks) {
        int kk = ks * 32;
        v8bf alo = *(const v8bf*)(arowBase + kk);
        v8bf ahi = *(const v8bf*)(arowBase + kk + 16);
        v16bf a = SHUF16(alo, ahi);
#pragma unroll
        for (int j = 0; j < 4; ++j) {
            int n = c0 + 16 * j + lm;
            const __bf16* bcol = BT + (size_t)n * 256 + kk + 8 * half;
            v16bf b = SHUF16(*(const v8bf*)bcol, *(const v8bf*)(bcol + 16));
            acc[j] = __builtin_amdgcn_wmma_f32_16x16x32_bf16(
                false, a, false, b, (short)0, acc[j], false, false);
        }
    }
#pragma unroll
    for (int j = 0; j < 4; ++j) {
        int n = c0 + 16 * j + lm;
        float bn = bias[n];
#pragma unroll
        for (int r = 0; r < 8; ++r) {
            int row = r0 + r + 8 * half;
            if (row < M) Out[(size_t)row * 128 + n] = acc[j][r] + bn;
        }
    }
}

// ---------------------------------------------------------------------------
// 6) per-column sum / sum-of-squares -> stats[0..127]=sum, stats[128..255]=sumsq
// ---------------------------------------------------------------------------
__global__ __launch_bounds__(256) void k_stats(const float* __restrict__ hn,
                                               float* __restrict__ stats, int Nn) {
    __shared__ float sA[256], sB[256];
    int tid = threadIdx.x, c = tid & 127, half = tid >> 7;
    float s = 0.f, s2 = 0.f;
    int r0 = blockIdx.x * 256;
    int rend = r0 + 256; if (rend > Nn) rend = Nn;
    for (int r = r0 + half; r < rend; r += 2) {
        float v = hn[(size_t)r * 128 + c];
        s += v; s2 += v * v;
    }
    sA[tid] = s; sB[tid] = s2;
    __syncthreads();
    if (half == 0) {
        atomicAdd(stats + c,       sA[tid] + sA[tid + 128]);
        atomicAdd(stats + 128 + c, sB[tid] + sB[tid + 128]);
    }
}

// ---------------------------------------------------------------------------
// 7) BatchNorm apply (+ optional ReLU): out = g*(x-mean)*rsqrt(var+eps)+b
// ---------------------------------------------------------------------------
__global__ void k_apply(const float* __restrict__ hn, const float* __restrict__ stats,
                        const float* __restrict__ gamma, const float* __restrict__ beta,
                        float* __restrict__ out, int Nn, int do_relu) {
    int idx = blockIdx.x * blockDim.x + threadIdx.x;   // Nn*32 float4 slots
    int node = idx >> 5, c4 = idx & 31;
    if (node >= Nn) return;
    float4 v = ((const float4*)hn)[(size_t)node * 32 + c4];
    float t[4] = { v.x, v.y, v.z, v.w };
    float invN = 1.0f / (float)Nn;
    float o[4];
#pragma unroll
    for (int j = 0; j < 4; ++j) {
        int c = c4 * 4 + j;
        float mean = stats[c] * invN;
        float var  = stats[128 + c] * invN - mean * mean;
        float inv  = rsqrtf(var + 1e-5f);
        float r = gamma[c] * (t[j] - mean) * inv + beta[c];
        if (do_relu) r = fmaxf(r, 0.0f);
        o[j] = r;
    }
    float4 ov; ov.x = o[0]; ov.y = o[1]; ov.z = o[2]; ov.w = o[3];
    ((float4*)out)[(size_t)node * 32 + c4] = ov;
}

// ---------------------------------------------------------------------------
// Launcher
// ---------------------------------------------------------------------------
extern "C" void kernel_launch(void* const* d_in, const int* in_sizes, int n_in,
                              void* d_out, int out_size, void* d_ws, size_t ws_size,
                              hipStream_t stream) {
    const int*   x          = (const int*)  d_in[0];
    const int*   edge_index = (const int*)  d_in[1];
    const int*   edge_attr  = (const int*)  d_in[2];
    const float* x_emb1     = (const float*)d_in[3];
    const float* x_emb2     = (const float*)d_in[4];
    const float* edge_emb1  = (const float*)d_in[5];   // [5,6,128]
    const float* edge_emb2  = (const float*)d_in[6];   // [5,3,128]
    const float* w1         = (const float*)d_in[7];   // [5,128,256]
    const float* b1         = (const float*)d_in[8];   // [5,256]
    const float* w2         = (const float*)d_in[9];   // [5,256,128]
    const float* b2         = (const float*)d_in[10];  // [5,128]
    const float* bn_gamma   = (const float*)d_in[11];  // [5,128]
    const float* bn_beta    = (const float*)d_in[12];  // [5,128]

    const int Nn = in_sizes[0] / 2;     // 200000
    const int E  = in_sizes[1] / 2;     // 2000000
    const int L  = 5;

    // workspace layout
    char* ws = (char*)d_ws;
    const size_t szNF = (size_t)Nn * 128 * sizeof(float);   // 102.4 MB
    float*  h     = (float*) (ws + 0 * szNF);
    float*  agg   = (float*) (ws + 1 * szNF);
    float*  hn    = (float*) (ws + 2 * szNF);
    __bf16* hid   = (__bf16*)(ws + 3 * szNF);               // N x 256 bf16 (= szNF bytes)
    __bf16* w1bT  = (__bf16*)(ws + 4 * szNF);               // [L][256][128] bf16
    __bf16* w2bT  = (__bf16*)((char*)w1bT + (size_t)L * 128 * 256 * 2);  // [L][128][256]
    float*  stats = (float*) ((char*)w2bT + (size_t)L * 256 * 128 * 2);

    // convert + transpose weights to bf16 (N-major)
    {
        long n1 = (long)L * 128 * 256;
        long n2 = (long)L * 256 * 128;
        k_cvtT<<<(int)((n1 + 255) / 256), 256, 0, stream>>>(w1, w1bT, 128, 256, n1);
        k_cvtT<<<(int)((n2 + 255) / 256), 256, 0, stream>>>(w2, w2bT, 256, 128, n2);
    }

    // node embeddings
    {
        long nthr = (long)Nn * 32;
        k_embed<<<(int)((nthr + 255) / 256), 256, 0, stream>>>(x, x_emb1, x_emb2, h, Nn);
    }

    for (int l = 0; l < L; ++l) {
        long n4 = (long)Nn * 32;
        k_zero<<<(int)((n4 + 255) / 256), 256, 0, stream>>>(
            (float4*)agg, (float4*)stats, n4);

        long totalE = (long)E + Nn;
        k_msg<<<(int)((totalE + 7) / 8), 256, 0, stream>>>(
            edge_index, edge_attr, h,
            edge_emb1 + (size_t)l * 6 * 128,
            edge_emb2 + (size_t)l * 3 * 128,
            agg, E, Nn);

        k_gemm1<<<(Nn + 31) / 32, 256, 0, stream>>>(
            agg, w1bT + (size_t)l * 256 * 128, b1 + (size_t)l * 256, hid, Nn);

        k_gemm2<<<(Nn + 63) / 64, 256, 0, stream>>>(
            hid, w2bT + (size_t)l * 128 * 256, b2 + (size_t)l * 128, hn, Nn);

        k_stats<<<(Nn + 255) / 256, 256, 0, stream>>>(hn, stats, Nn);

        float* dst = (l == L - 1) ? (float*)d_out : h;
        long nthr = (long)Nn * 32;
        k_apply<<<(int)((nthr + 255) / 256), 256, 0, stream>>>(
            hn, stats, bn_gamma + (size_t)l * 128, bn_beta + (size_t)l * 128,
            dst, Nn, (l < L - 1) ? 1 : 0);
    }
}
